// PolymorphicDenseBase_50517405336015
// MI455X (gfx1250) — matrix operand order repaired and checked by reading, not compiled, not run on hardware
//
#include <hip/hip_runtime.h>
#include <hip/hip_bf16.h>
#include <cstdint>

typedef __bf16 v16bf __attribute__((ext_vector_type(16)));
typedef __bf16 v8bf  __attribute__((ext_vector_type(8)));
typedef float  v8f   __attribute__((ext_vector_type(8)));
typedef unsigned int v4u __attribute__((ext_vector_type(4)));
typedef int v4i __attribute__((ext_vector_type(4)));
typedef int v8i __attribute__((ext_vector_type(8)));

#define BDIM    16384
#define KEYDIM  64
#define DDIM    128
#define UDIM    128
#define NMODES  32
#define BT      64          // batch rows per block
#define THREADS 256         // 8 waves (wave32)

// padded LDS strides (bf16 elems): fragment chunks stay 16B aligned, banks spread
#define KSTR 136            // kernels[m]^T tile [u][d]  (8-half pad == 4 DWORDs / 64 DWORDs)
#define XSTR 136            // x tile [r][d]
#define SSTR 40             // sim bf16 [r][m]
#define BSTR 40             // biases^T [u][m]

#define KTILE_ELEMS (DDIM * UDIM)                       // 16384 bf16 per mode
#define KTILE_LDS   (UDIM * KSTR)                       // padded tile span in bf16

#define CAT16(lo, hi) __builtin_shufflevector(lo, hi, 0,1,2,3,4,5,6,7,8,9,10,11,12,13,14,15)

// ---------------------------------------------------------------------------
// TDM: issue a 32KB contiguous bf16 tile load global->LDS with hardware
// padding of 4 DWORDs every 64 DWORDs (reproduces the KSTR=136 layout).
// Descriptor per CDNA5 ISA ch.10.8 (D# groups 0/1; groups 2/3 zero => <=2D).
// ---------------------------------------------------------------------------
__device__ __forceinline__ void tdm_tile_load(uint32_t lds_byte_addr, const void* gptr) {
    uint64_t ga = (uint64_t)(uintptr_t)gptr;
    v4u g0 = { 1u,                                   // count=1, user mode
               lds_byte_addr,                        // lds_addr
               (uint32_t)ga,                         // global_addr[31:0]
               (((uint32_t)(ga >> 32)) & 0x01FFFFFFu) | 0x80000000u }; // ga[56:32] | type=2
    v8i g1 = { (1 << 16)      // data_size = 2 bytes (bf16)
             | (1 << 20)      // pad_enable
             | (5 << 22)      // pad_interval: 64 DWORDs (256B) between pads
             | (3 << 25),     // pad_amount : 4 DWORDs (16B)
               (int)(16384u << 16),   // tensor_dim0[15:0]=16384 in [31:16]
               (int)(1u << 16),       // tensor_dim0 hi=0; tensor_dim1[15:0]=1
               (int)(16384u << 16),   // tensor_dim1 hi=0; tile_dim0=16384
               1,                     // tile_dim1=1, tile_dim2=0
               16384,                 // tensor_dim0_stride lo
               0, 0 };
    v4i z4 = { 0, 0, 0, 0 };
#if __clang_major__ >= 23
    v8i z8 = { 0, 0, 0, 0, 0, 0, 0, 0 };
    __builtin_amdgcn_tensor_load_to_lds(g0, g1, z4, z4, z8, 0);
#else
    __builtin_amdgcn_tensor_load_to_lds(g0, g1, z4, z4, 0);
#endif
}

// ---------------------------------------------------------------------------
// Pre-pass: kernels [m][d][u] f32  ->  kbfT [m][u][d] bf16 (contiguous rows)
// ---------------------------------------------------------------------------
__global__ __launch_bounds__(256)
void convert_kernels_bf16T(const float* __restrict__ kernels, __bf16* __restrict__ kbfT) {
    int e = blockIdx.x * 256 + threadIdx.x;       // over NMODES*DDIM*UDIM
    int m = e >> 14;
    int r = e & 16383;
    int u = r >> 7;
    int d = r & 127;
    kbfT[e] = (__bf16)kernels[(m << 14) + d * UDIM + u];
}

// ---------------------------------------------------------------------------
// Fused kernel. USE_TDM: kernels pre-converted in d_ws, streamed by the
// Tensor Data Mover into a double-buffered LDS tile, overlapped with WMMA.
// ---------------------------------------------------------------------------
template <bool USE_TDM>
__global__ __launch_bounds__(THREADS, 1)
void poly_dense_wmma(const float* __restrict__ key,
                     const float* __restrict__ x,
                     const float* __restrict__ sens,
                     const float* __restrict__ keys_map,
                     const float* __restrict__ kernels,
                     const float* __restrict__ biases,
                     const __bf16* __restrict__ kbfT,
                     float* __restrict__ out)
{
    __shared__ __align__(32) __bf16 ldsK[(USE_TDM ? 2 : 1) * KTILE_LDS]; // 34816 or 69632 B
    __shared__ __align__(32) __bf16 ldsX[BT * XSTR];                     // 17408 B
    __shared__ __align__(32) float  ldsSimF[BT * NMODES];                //  8192 B
    __shared__ __align__(32) __bf16 ldsSimH[BT * SSTR];                  //  5120 B
    __shared__ __align__(32) __bf16 ldsBT[UDIM * BSTR];                  // 10240 B
    __shared__ __align__(32) float  ldsKM[NMODES * KEYDIM];              //  8192 B

    const int tid  = threadIdx.x;
    const int lane = tid & 31;
    const int wave = tid >> 5;
    const int rowBase = blockIdx.x * BT;

    // kick the DMA for mode 0 immediately: streams in while we do phase 1
    if (USE_TDM && wave == 0) {
        tdm_tile_load((uint32_t)(uintptr_t)(void*)&ldsK[0], kbfT);
    }

    // ---- stage keys_map (f32), x tile (->bf16), biases^T (->bf16) ----
    #pragma unroll
    for (int i = 0; i < (NMODES * KEYDIM) / THREADS; ++i) {
        int e = tid + i * THREADS;
        ldsKM[e] = keys_map[e];
    }
    #pragma unroll
    for (int i = 0; i < (BT * DDIM) / THREADS; ++i) {
        int e = tid + i * THREADS;
        int r = e >> 7, c = e & 127;
        ldsX[r * XSTR + c] = (__bf16)x[(rowBase + r) * DDIM + c];
    }
    #pragma unroll
    for (int i = 0; i < (NMODES * UDIM) / THREADS; ++i) {
        int e = tid + i * THREADS;
        int m = e >> 7, u = e & 127;
        ldsBT[u * BSTR + m] = (__bf16)biases[m * UDIM + u];
    }
    __syncthreads();

    // ---- phase 1: distances + logits.  thread -> (row = tid/4, 8 modes) ----
    {
        const int bl = tid >> 2;
        const int m0 = (tid & 3) * 8;
        float accd[8];
        #pragma unroll
        for (int j = 0; j < 8; ++j) accd[j] = 0.0f;
        const float* kr = key + (size_t)(rowBase + bl) * KEYDIM;
        for (int k = 0; k < KEYDIM; ++k) {
            float kv = kr[k];
            #pragma unroll
            for (int j = 0; j < 8; ++j) {
                float d = kv - ldsKM[(m0 + j) * KEYDIM + k];
                accd[j] += d * d;
            }
        }
        #pragma unroll
        for (int j = 0; j < 8; ++j) {
            float dist = sqrtf(accd[j]);
            ldsSimF[bl * NMODES + m0 + j] = sens[m0 + j] / (dist + 1.0f);
        }
    }
    __syncthreads();

    // ---- softmax over modes (one thread per row), write f32 + bf16 ----
    if (tid < BT) {
        float mx = -3.4e38f;
        for (int m = 0; m < NMODES; ++m) mx = fmaxf(mx, ldsSimF[tid * NMODES + m]);
        float s = 0.0f;
        for (int m = 0; m < NMODES; ++m) s += __expf(ldsSimF[tid * NMODES + m] - mx);
        float inv = 1.0f / s;
        for (int m = 0; m < NMODES; ++m) {
            float v = __expf(ldsSimF[tid * NMODES + m] - mx) * inv;
            ldsSimF[tid * NMODES + m] = v;
            ldsSimH[tid * SSTR + m]   = (__bf16)v;
        }
    }
    __syncthreads();

    // ---- GEMM: out_tile[64x128] = sum_m sim[:,m]*(x @ K_m)  + sim @ biases ----
    const int wr   = wave & 3;           // row strip (16 rows)
    const int wc   = wave >> 2;          // col half (64 cols)
    const int mrow = lane & 15;
    const int hig  = lane >> 4;          // 0 or 1
    const int arow = wr * 16 + mrow;     // local batch row for A fragments
    const int akb  = hig * 8;            // A K-chunk base (ISA 16-bit A layout)
    const int bkb  = hig * 8;            // B K-chunk base

    v8f acc[4] = {};

    for (int m = 0; m < NMODES; ++m) {
        const __bf16* kcur;
        if (USE_TDM) {
            if (wave == 0) {
                __builtin_amdgcn_s_wait_tensorcnt(0);    // tile m landed in LDS
            }
            __syncthreads();                             // publish to all waves
            if (wave == 0 && m + 1 < NMODES) {           // stream m+1 during compute of m
                tdm_tile_load((uint32_t)(uintptr_t)(void*)&ldsK[((m + 1) & 1) * KTILE_LDS],
                              kbfT + (size_t)(m + 1) * KTILE_ELEMS);
            }
            kcur = &ldsK[(m & 1) * KTILE_LDS];
        } else {
            __syncthreads();                             // previous ldsK consumers done
            const float* km = kernels + (size_t)m * DDIM * UDIM;
            if (m + 1 < NMODES)
                __builtin_prefetch(kernels + (size_t)(m + 1) * DDIM * UDIM + tid * 16, 0, 0);
            #pragma unroll
            for (int i = 0; i < (DDIM * UDIM) / THREADS; ++i) {
                int e = tid + i * THREADS;
                int d = e >> 7, u = e & 127;
                ldsK[u * KSTR + d] = (__bf16)km[e];      // transpose to [u][d]
            }
            __syncthreads();
            kcur = ldsK;
        }

        const __bf16 s = (__bf16)ldsSimF[arow * NMODES + m];

        #pragma unroll
        for (int ks = 0; ks < 4; ++ks) {
            const __bf16* ap = &ldsX[arow * XSTR + ks * 32 + akb];
            v8bf alo = *(const v8bf*)ap;
            v8bf ahi = *(const v8bf*)(ap + 16);
            v16bf a = CAT16(alo, ahi);
            a = a * s;                                   // per-row sim scaling
            #pragma unroll
            for (int ct = 0; ct < 4; ++ct) {
                const __bf16* bp = &kcur[(wc * 64 + ct * 16 + mrow) * KSTR + ks * 32 + bkb];
                v8bf blo = *(const v8bf*)bp;
                v8bf bhi = *(const v8bf*)(bp + 16);
                v16bf b = CAT16(blo, bhi);
                acc[ct] = __builtin_amdgcn_wmma_f32_16x16x32_bf16(
                    false, a, false, b, (short)0, acc[ct], false, false);
            }
        }
    }

    // ---- bias fold: one extra k-step, A = sim (K=32), B = biases^T ----
    {
        const __bf16* ap = &ldsSimH[arow * SSTR + akb];
        v8bf alo = *(const v8bf*)ap;
        v8bf ahi = *(const v8bf*)(ap + 16);
        v16bf a = CAT16(alo, ahi);
        #pragma unroll
        for (int ct = 0; ct < 4; ++ct) {
            const __bf16* bp = &ldsBT[(wc * 64 + ct * 16 + mrow) * BSTR + bkb];
            v8bf blo = *(const v8bf*)bp;
            v8bf bhi = *(const v8bf*)(bp + 16);
            v16bf b = CAT16(blo, bhi);
            acc[ct] = __builtin_amdgcn_wmma_f32_16x16x32_bf16(
                false, a, false, b, (short)0, acc[ct], false, false);
        }
    }

    // ---- store: out = acc / MODES  (C layout: VGPR v -> row v + hig*8) ----
    const float inv32 = 1.0f / (float)NMODES;
    #pragma unroll
    for (int ct = 0; ct < 4; ++ct) {
        const int col = wc * 64 + ct * 16 + mrow;
        #pragma unroll
        for (int v = 0; v < 8; ++v) {
            int row = rowBase + wr * 16 + v + hig * 8;
            out[(size_t)row * UDIM + col] = acc[ct][v] * inv32;
        }
    }
}

extern "C" void kernel_launch(void* const* d_in, const int* in_sizes, int n_in,
                              void* d_out, int out_size, void* d_ws, size_t ws_size,
                              hipStream_t stream) {
    const float* key      = (const float*)d_in[0];
    const float* x        = (const float*)d_in[1];
    const float* sens     = (const float*)d_in[2];
    const float* keys_map = (const float*)d_in[3];
    const float* kernels  = (const float*)d_in[4];
    const float* biases   = (const float*)d_in[5];
    float* out = (float*)d_out;

    dim3 grid(BDIM / BT);     // 256 blocks
    dim3 block(THREADS);      // 8 waves

    const size_t kbf_bytes = (size_t)NMODES * DDIM * UDIM * sizeof(__bf16);  // 1 MB
    if (d_ws != nullptr && ws_size >= kbf_bytes) {
        __bf16* kbfT = (__bf16*)d_ws;
        convert_kernels_bf16T<<<(NMODES * DDIM * UDIM) / 256, 256, 0, stream>>>(kernels, kbfT);
        poly_dense_wmma<true><<<grid, block, 0, stream>>>(
            key, x, sens, keys_map, kernels, biases, kbfT, out);
    } else {
        poly_dense_wmma<false><<<grid, block, 0, stream>>>(
            key, x, sens, keys_map, kernels, biases, nullptr, out);
    }
}
